// GCN_24661702214226
// MI455X (gfx1250) — compile-verified, matrix operand order checked by reading, not run on hardware
//
#include <hip/hip_runtime.h>

#define N_NODES 100000
#define N_EDGES 1600000
#define F_IN 128
#define HID 128
#define N_CLS 40
#define N_CLS_PAD 48

static_assert(N_NODES % 16 == 0, "exact 16-row tiling assumed (no store guards)");

typedef __bf16 bf16_t;
typedef __attribute__((ext_vector_type(16))) __bf16 v16bf;
typedef __attribute__((ext_vector_type(8)))  float  v8f;

// ---- gfx1250 async global->LDS DMA (guarded: falls back to sync staging) ----
#if defined(__has_builtin)
#if __has_builtin(__builtin_amdgcn_global_load_async_to_lds_b128)
#define USE_ASYNC_LDS 1
#endif
#endif

// b128 async payload type per clang's builtin prototype: int4 vector pointer.
typedef __attribute__((__vector_size__(16))) int v4i;
typedef __attribute__((address_space(3))) v4i v4i_lds;
// clang forbids direct cross-AS pointer casts; go through integers.
// generic LDS pointer: low 32 bits are the LDS byte offset.
#define TO_LDS_V4I(p) ((v4i_lds*)(unsigned int)(uintptr_t)(p))

__device__ __forceinline__ void wait_async0() {
#if defined(__has_builtin)
#if __has_builtin(__builtin_amdgcn_s_wait_asynccnt)
    __builtin_amdgcn_s_wait_asynccnt(0);
    return;
#endif
#endif
    asm volatile("s_wait_asynccnt 0x0" ::: "memory");
}

// ---------------------------------------------------------------- utilities

__global__ void k_fill(float* __restrict__ p, float v, long long n) {
    long long i = (long long)blockIdx.x * blockDim.x + threadIdx.x;
    if (i < n) p[i] = v;
}

// deg[d] += 1 per incoming edge (deg pre-filled with 1.0 for the self-loop)
__global__ void k_degree(const long long* __restrict__ dst, float* __restrict__ deg, int nE) {
    int e = blockIdx.x * blockDim.x + threadIdx.x;
    if (e < nE) atomicAdd(&deg[(int)dst[e]], 1.0f);
}

__global__ void k_rsqrt(float* __restrict__ d, int n) {
    int i = blockIdx.x * blockDim.x + threadIdx.x;
    if (i < n) d[i] = rsqrtf(d[i]);
}

// WT[n*128+k] = (n < nIn) ? (bf16)W[k*nIn + n] : 0   (transpose + pad + cvt)
__global__ void k_convert_wT(const float* __restrict__ W, bf16_t* __restrict__ WT,
                             int nIn, int nOut) {
    int i = blockIdx.x * blockDim.x + threadIdx.x;
    if (i >= nOut * 128) return;
    int n = i >> 7, k = i & 127;
    WT[i] = (n < nIn) ? (bf16_t)W[k * nIn + n] : (bf16_t)0.0f;
}

// ------------------------------------------------------------------- GEMM
// C[M x (16*NCT)] = A[M x 128] * B, with BT = bf16 B-transposed [16*NCT x 128].
// One block = 16 output rows; wave w computes the 16x16 tile at columns w*16.
// A tile staged in LDS: async-DMA f32 (cvt->bf16 during fragment build) when
// the gfx1250 async builtin exists, else VGPR round-trip with cvt on store.
#define AS_F32 132   // f32 row stride: 128 + 4 pad (528B) breaks bank alignment
#define AS_BF  136   // bf16 row stride: 128 + 8 pad (272B)

template<int NCT>
__global__ __launch_bounds__(32 * NCT)
void k_gemm_bf16(const float* __restrict__ A, const bf16_t* __restrict__ BT,
                 float* __restrict__ C, int M) {
    const int lane = threadIdx.x;          // 0..31
    const int wv   = threadIdx.y;          // column tile
    const int tid  = wv * 32 + lane;
    const int rowBase = blockIdx.x * 16;   // grid is exactly M/16
    const int mr    = lane & 15;
    const int kHalf = (lane >> 4) << 4;

#if USE_ASYNC_LDS
    __shared__ __align__(16) float Af[16 * AS_F32];
    // 16 rows x 32 16-byte chunks, one async DMA per chunk (ASYNCcnt-tracked)
    for (int ch = tid; ch < 16 * 32; ch += 32 * NCT) {
        int r = ch >> 5, c4 = (ch & 31) << 2;
        __builtin_amdgcn_global_load_async_to_lds_b128(
            (v4i*)(A + (long long)(rowBase + r) * 128 + c4),
            TO_LDS_V4I(Af + r * AS_F32 + c4), 0, 0);
    }
    wait_async0();
    __syncthreads();
    const float* arow = Af + mr * AS_F32 + kHalf;
#else
    __shared__ __align__(32) bf16_t Abf[16 * AS_BF];
    for (int i = tid; i < 16 * 128; i += 32 * NCT) {
        int r = i >> 7, c = i & 127;
        Abf[r * AS_BF + c] = (bf16_t)A[(long long)(rowBase + r) * 128 + c];
    }
    __syncthreads();
    const bf16_t* arow = Abf + mr * AS_BF + kHalf;
#endif

    v8f acc = {};
    const bf16_t* brow = BT + ((wv * 16 + mr) * 128 + kHalf);
#pragma unroll
    for (int ks = 0; ks < 4; ++ks) {       // K = 128 = 4 x 32
        v16bf a;
#if USE_ASYNC_LDS
        const float* p = arow + ks * 32;
#pragma unroll
        for (int j = 0; j < 16; ++j) a[j] = (bf16_t)p[j];
#else
        a = *(const v16bf*)(arow + ks * 32);
#endif
        v16bf b = *(const v16bf*)(brow + ks * 32);
        acc = __builtin_amdgcn_wmma_f32_16x16x32_bf16(
            false, a, false, b, (short)0, acc, false, false);
    }

    // D layout: VGPR v -> row v (lanes 0-15) / 8+v (lanes 16-31), col = lane&15
    const int ldc  = NCT * 16;
    const int ccol = wv * 16 + mr;
    const long long rtop = rowBase + ((lane >> 4) << 3);
#pragma unroll
    for (int v = 0; v < 8; ++v)
        C[(rtop + v) * ldc + ccol] = acc[v];
}

// -------------------------------------------------------------- aggregation
// AGG[dst, f] += H[src, f] * dinv[src]*dinv[dst]; one thread per edge-quad:
// b128 gather + 4 f32 atomics, 4x fewer int64 index / dinv loads than scalar.
template<int F>
__global__ __launch_bounds__(256)
void k_aggregate(const float* __restrict__ H, const long long* __restrict__ src,
                 const long long* __restrict__ dst, const float* __restrict__ dinv,
                 float* __restrict__ AGG) {
    constexpr int Q = F / 4;
    long long t = (long long)blockIdx.x * blockDim.x + threadIdx.x;
    if (t >= (long long)N_EDGES * Q) return;
    int e = (int)(t / Q);
    int q = (int)(t - (long long)e * Q) << 2;
    int s = (int)src[e], d = (int)dst[e];
    float nrm = dinv[s] * dinv[d];
    const float4 h = *(const float4*)(H + (long long)s * F + q);
    float* a = AGG + (long long)d * F + q;
    atomicAdd(a + 0, h.x * nrm);
    atomicAdd(a + 1, h.y * nrm);
    atomicAdd(a + 2, h.z * nrm);
    atomicAdd(a + 3, h.w * nrm);
}

// H1 = relu(AGG + H * dinv^2 + b1), written into AGG in place
__global__ void k_finalize1(float* __restrict__ AGG, const float* __restrict__ H,
                            const float* __restrict__ dinv, const float* __restrict__ b1) {
    long long i = (long long)blockIdx.x * blockDim.x + threadIdx.x;
    if (i >= (long long)N_NODES * HID) return;
    int node = (int)(i >> 7), f = (int)(i & 127);
    float di = dinv[node];
    float v = AGG[i] + H[i] * di * di + b1[f];
    AGG[i] = v > 0.0f ? v : 0.0f;
}

// logits = AGG2 + H2*dinv^2 + b2 ; out = log_softmax(logits)
__global__ void k_logsoftmax(const float* __restrict__ AGG2, const float* __restrict__ H2,
                             const float* __restrict__ dinv, const float* __restrict__ b2,
                             float* __restrict__ out) {
    int n = blockIdx.x * blockDim.x + threadIdx.x;
    if (n >= N_NODES) return;
    float dd = dinv[n]; dd *= dd;
    float l[N_CLS];
    float mx = -3.4e38f;
    for (int c = 0; c < N_CLS; ++c) {
        float v = AGG2[(long long)n * N_CLS_PAD + c]
                + H2[(long long)n * N_CLS_PAD + c] * dd + b2[c];
        l[c] = v;
        mx = fmaxf(mx, v);
    }
    float s = 0.0f;
    for (int c = 0; c < N_CLS; ++c) s += __expf(l[c] - mx);
    float lg = mx + __logf(s);
    for (int c = 0; c < N_CLS; ++c) out[(long long)n * N_CLS + c] = l[c] - lg;
}

// ------------------------------------------------------------------ driver

static inline int cdiv(long long a, long long b) { return (int)((a + b - 1) / b); }

extern "C" void kernel_launch(void* const* d_in, const int* in_sizes, int n_in,
                              void* d_out, int out_size, void* d_ws, size_t ws_size,
                              hipStream_t stream) {
    const float*     X   = (const float*)d_in[0];
    const long long* EI  = (const long long*)d_in[1];   // int64 [2, E]
    const float*     W1  = (const float*)d_in[2];
    const float*     b1  = (const float*)d_in[3];
    const float*     W2  = (const float*)d_in[4];
    const float*     b2  = (const float*)d_in[5];
    const long long* src = EI;
    const long long* dst = EI + N_EDGES;
    float* out = (float*)d_out;

    // workspace layout (all offsets 256B-aligned)
    char* ws = (char*)d_ws;
    float*  dinv = (float*)(ws);                                      // 400 KB
    float*  H    = (float*)(ws + 512 * 1024);                         // 51.2 MB
    float*  AGG  = (float*)(ws + 512 * 1024 + 51200000);              // 51.2 MB
    bf16_t* W1bf = (bf16_t*)(ws + 512 * 1024 + 2 * 51200000);         // 32 KB
    bf16_t* W2bf = (bf16_t*)(ws + 512 * 1024 + 2 * 51200000 + 32768); // 12 KB
    float*  H2   = H;                                    // reuse H region (19.2 MB)
    float*  AGG2 = H + (long long)N_NODES * N_CLS_PAD;   // + 19.2 MB, still inside H

    // degree -> dinv
    k_fill<<<cdiv(N_NODES, 256), 256, 0, stream>>>(dinv, 1.0f, N_NODES);
    k_fill<<<cdiv((long long)N_NODES * HID, 256), 256, 0, stream>>>(AGG, 0.0f, (long long)N_NODES * HID);
    k_degree<<<cdiv(N_EDGES, 256), 256, 0, stream>>>(dst, dinv, N_EDGES);
    k_rsqrt<<<cdiv(N_NODES, 256), 256, 0, stream>>>(dinv, N_NODES);

    // weight prep: transpose + pad + bf16
    k_convert_wT<<<cdiv(HID * 128, 256), 256, 0, stream>>>(W1, W1bf, HID, HID);
    k_convert_wT<<<cdiv(N_CLS_PAD * 128, 256), 256, 0, stream>>>(W2, W2bf, N_CLS, N_CLS_PAD);

    // layer 1: H = X @ W1 (WMMA), scatter, fused bias+self-loop+relu
    k_gemm_bf16<8><<<N_NODES / 16, dim3(32, 8), 0, stream>>>(X, W1bf, H, N_NODES);
    k_aggregate<128><<<cdiv((long long)N_EDGES * 32, 256), 256, 0, stream>>>(H, src, dst, dinv, AGG);
    k_finalize1<<<cdiv((long long)N_NODES * HID, 256), 256, 0, stream>>>(AGG, H, dinv, b1);

    // layer 2: H2 = H1 @ W2 (padded to 48 cols), scatter, fused log_softmax
    k_gemm_bf16<3><<<N_NODES / 16, dim3(32, 3), 0, stream>>>(AGG, W2bf, H2, N_NODES);
    k_fill<<<cdiv((long long)N_NODES * N_CLS_PAD, 256), 256, 0, stream>>>(AGG2, 0.0f, (long long)N_NODES * N_CLS_PAD);
    k_aggregate<48><<<cdiv((long long)N_EDGES * 12, 256), 256, 0, stream>>>(H2, src, dst, dinv, AGG2);
    k_logsoftmax<<<cdiv(N_NODES, 256), 256, 0, stream>>>(AGG2, H2, dinv, b2, out);

    (void)in_sizes; (void)n_in; (void)out_size; (void)ws_size;
}
